// GRU_D_ATTN_V1_65420941853080
// MI455X (gfx1250) — compile-verified, matrix operand order checked by reading, not run on hardware
//
#include <hip/hip_runtime.h>
#include <hip/hip_bf16.h>
#include <math.h>

// Problem sizes (fixed by the reference)
#define Bsz 1024
#define Tw  512
#define Fin 17
#define Hd  256
#define H3  768

// ---- WMMA types (CDNA5 gfx1250, wave32) ----
typedef __bf16 bf16r;
typedef bf16r v16bf __attribute__((ext_vector_type(16)));
typedef float v8f   __attribute__((ext_vector_type(8)));

union Frag { uint4 u[2]; v16bf v; };

// Load one 16-lane x 16-element bf16 fragment for V_WMMA_F32_16X16X32_BF16.
// Per-lane layout (ISA 7.12.2, 16-bit A 16x32 / B 32x16):
//   lane l (l<16): row = l,    K-chunks [k0+0..7]  and [k0+16..23]
//   lane l (>=16): row = l-16, K-chunks [k0+8..15] and [k0+24..31]
// 'row' is the M-row (A operand) or weight row j == N-column (B operand).
__device__ __forceinline__ v16bf frag_ld(const __hip_bfloat16* base, size_t row,
                                         int ld, int k0, int lane) {
  const __hip_bfloat16* p = base + row * (size_t)ld + (size_t)(k0 + ((lane >> 4) << 3));
  Frag f;
  f.u[0] = *(const uint4*)(p);
  f.u[1] = *(const uint4*)(p + 16);
  return f.v;
}

// ---------------------------------------------------------------------------
// Kernel 1: collapse init_W/init_b over the 128 hidden units (softmax only
// consumes the mean over k):  Wbar[i,t] = mean_k init_W[i,k,t], bbar[i] = mean_k init_b[i,k]
__global__ void k_wbar(const float* __restrict__ initW, const float* __restrict__ initb,
                       float* __restrict__ Wbar, float* __restrict__ bbar) {
  int idx = blockIdx.x * 256 + threadIdx.x;
  if (idx < Fin * Tw) {
    int i = idx / Tw, t = idx - i * Tw;
    float s = 0.f;
    for (int k = 0; k < 128; ++k) s += initW[((size_t)i * 128 + k) * Tw + t];
    Wbar[idx] = s * (1.f / 128.f);
  }
  if (idx < Fin) {
    float s = 0.f;
    for (int k = 0; k < 128; ++k) s += initb[idx * 128 + k];
    bbar[idx] = s * (1.f / 128.f);
  }
}

// ---------------------------------------------------------------------------
// Kernel 2: per-batch imputation + attention logits + softmax, emit bf16
// XT[b,t,0:17]=x_imputed*attn, XT[b,t,17:34]=m, XT[b,t,34:64]=0  (K padded to 64)
__global__ void k_attn(const float* __restrict__ x, const float* __restrict__ mm,
                       const float* __restrict__ d, const float* __restrict__ x_m,
                       const float* __restrict__ dec_w, const float* __restrict__ dec_b,
                       const float* __restrict__ Wbar, const float* __restrict__ bbar,
                       __hip_bfloat16* __restrict__ XT) {
  int b = blockIdx.x, tid = threadIdx.x;
  __shared__ float part[Fin][257];
  __shared__ float lg[Fin];
  __shared__ float attn[Fin];

  float acc[Fin];
#pragma unroll
  for (int i = 0; i < Fin; ++i) acc[i] = 0.f;

  size_t base = (size_t)b * Tw * Fin;
  for (int tt = tid; tt < Tw; tt += 256) {
    size_t o = base + (size_t)tt * Fin;
#pragma unroll
    for (int i = 0; i < Fin; ++i) {
      float xv = x[o + i], mv = mm[o + i], dv = d[o + i];
      float xd = __expf(-fmaxf(dv * dec_w[i] + dec_b[i], 0.f));
      float xi = mv * xv + (1.f - mv) * (xd * xv + (1.f - xd) * x_m[i]);
      acc[i] += xi * Wbar[i * Tw + tt];
    }
  }
#pragma unroll
  for (int i = 0; i < Fin; ++i) part[i][tid] = acc[i];
  __syncthreads();

  if (tid < Fin) {
    float s = bbar[tid];
    for (int q = 0; q < 256; ++q) s += part[tid][q];
    lg[tid] = s;
  }
  __syncthreads();
  if (tid == 0) {
    float mx = lg[0];
#pragma unroll
    for (int i = 1; i < Fin; ++i) mx = fmaxf(mx, lg[i]);
    float den = 0.f;
#pragma unroll
    for (int i = 0; i < Fin; ++i) den += __expf(lg[i] - mx);
    float inv = 1.f / den;
#pragma unroll
    for (int i = 0; i < Fin; ++i) attn[i] = __expf(lg[i] - mx) * inv;
  }
  __syncthreads();

  for (int tt = tid; tt < Tw; tt += 256) {
    size_t o = base + (size_t)tt * Fin;
    __hip_bfloat16* row = XT + ((size_t)b * Tw + tt) * 64;
#pragma unroll
    for (int i = 0; i < Fin; ++i) {
      float xv = x[o + i], mv = mm[o + i], dv = d[o + i];
      float xd = __expf(-fmaxf(dv * dec_w[i] + dec_b[i], 0.f));
      float xi = mv * xv + (1.f - mv) * (xd * xv + (1.f - xd) * x_m[i]);
      row[i]       = __float2bfloat16(xi * attn[i]);
      row[Fin + i] = __float2bfloat16(mv);
    }
#pragma unroll
    for (int c = 2 * Fin; c < 64; ++c) row[c] = __float2bfloat16(0.f);
  }
}

// ---------------------------------------------------------------------------
// Kernel 3: weight conversion to bf16 (W_ih padded K 34 -> 64)
__global__ void k_convw(const float* __restrict__ Wih, const float* __restrict__ Whh,
                        __hip_bfloat16* __restrict__ Wihp, __hip_bfloat16* __restrict__ Whhb) {
  int idx = blockIdx.x * 256 + threadIdx.x;
  const int n1 = H3 * 64;
  if (idx < n1) {
    int j = idx >> 6, c = idx & 63;
    float v = (c < 2 * Fin) ? Wih[j * (2 * Fin) + c] : 0.f;
    Wihp[idx] = __float2bfloat16(v);
  } else {
    int k = idx - n1;
    if (k < H3 * Hd) Whhb[k] = __float2bfloat16(Whh[k]);
  }
}

// ---------------------------------------------------------------------------
// Kernel 4: GRU-D scan. 64 workgroups x 16 batch rows, 512 serial steps,
// h resident in LDS, gh/gi via v_wmma_f32_16x16x32_bf16.
// Dynamic LDS layout (floats unless noted):
//   h[4096] | Grz[8192] | Gni[4096] | Gnh[4096] | hhf[4096] | bias[1792]
//   | hdw[4352] | hhb[4096 bf16] | llen[16 int]
#define SCAN_SMEM_BYTES ((4096 + 8192 + 4096 + 4096 + 4096 + 1792 + 4352) * 4 + 4096 * 2 + 64)

__global__ void __launch_bounds__(256) k_scan(
    const __hip_bfloat16* __restrict__ XT,
    const __hip_bfloat16* __restrict__ Wihp,
    const __hip_bfloat16* __restrict__ Whhb,
    const float* __restrict__ d,
    const float* __restrict__ hdW,
    const float* __restrict__ hdb,
    const float* __restrict__ bih,
    const float* __restrict__ bhh,
    const float* __restrict__ h0,
    const int* __restrict__ length,
    float* __restrict__ sel) {
  extern __shared__ float smem[];
  float* h    = smem;          // [16][256]
  float* Grz  = h + 4096;      // [16][512] pre-activation r|z (gi+gh)
  float* Gni  = Grz + 8192;    // [16][256] gi for n-gate
  float* Gnh  = Gni + 4096;    // [16][256] gh for n-gate
  float* hhf  = Gnh + 4096;    // [16][256] h_dec*h (f32, for z*hh)
  float* bias = hhf + 4096;    // hd_b[256] | b_ih[768] | b_hh[768]
  float* hdw  = bias + 1792;   // hd_W staged [256][17]
  __hip_bfloat16* hhb = (__hip_bfloat16*)(hdw + 4352);  // [16][256] bf16
  int* llen = (int*)(hhb + 4096);

  int tid = threadIdx.x;
  int b0 = blockIdx.x * 16;

  for (int q = tid; q < 4096; q += 256) h[q] = h0[(size_t)b0 * Hd + q];
  for (int q = tid; q < 1792; q += 256)
    bias[q] = (q < 256) ? hdb[q] : (q < 1024 ? bih[q - 256] : bhh[q - 1024]);
  for (int q = tid; q < Hd * Fin; q += 256) hdw[q] = hdW[q];
  if (tid < 16) llen[tid] = length[b0 + tid];
  __syncthreads();

  int lane = tid & 31;
  int w = __builtin_amdgcn_readfirstlane(tid >> 5);  // scalar wave id
  int mrow = lane & 15;                 // WMMA M-row / N-col index for this lane
  int mthr = tid >> 4;                  // row for elementwise phases
  int cthr = (tid & 15) << 4;           // 16-col slab for elementwise phases

  // Laundered zero offset: redefined each iteration via empty asm so the
  // (t-invariant) weight fragment loads cannot be hoisted out of the scan
  // loop and spilled. Adding an int to the typed pointer keeps provenance,
  // so the loads stay global_load (not flat_load).
  int zoff = 0;

  for (int t = 0; t < Tw; ++t) {
    // ---- phase 1: hh = exp(-relu(d_t @ hd_W^T + hd_b)) * h (VALU, K=17) ----
    {
      const float* drow = d + ((size_t)(b0 + mthr) * Tw + t) * Fin;
      float dv[Fin];
#pragma unroll
      for (int i = 0; i < Fin; ++i) dv[i] = drow[i];
#pragma unroll 4
      for (int jj = 0; jj < 16; ++jj) {
        int j = cthr + jj;
        float a = bias[j];
        const float* wr = hdw + j * Fin;
#pragma unroll
        for (int i = 0; i < Fin; ++i) a += dv[i] * wr[i];
        float hv = __expf(-fmaxf(a, 0.f)) * h[mthr * 256 + j];
        hhf[mthr * 256 + j] = hv;
        hhb[mthr * 256 + j] = __float2bfloat16(hv);
      }
    }
    __syncthreads();

    // ---- phase 2: gi (K=64) and gh (K=256) via WMMA; wave w owns N-subtiles 6w..6w+5 ----
    {
      asm volatile("" : "+s"(zoff));
      const __hip_bfloat16* WhhT = Whhb + zoff;
      const __hip_bfloat16* WihT = Wihp + zoff;

      // A-fragments are shared by all 6 subtiles: load once per step.
      size_t xtrow = (size_t)(b0 + mrow) * Tw + t;
      v16bf Ai[2], Ah[8];
#pragma unroll
      for (int kk = 0; kk < 2; ++kk) Ai[kk] = frag_ld(XT, xtrow, 64, kk * 32, lane);
#pragma unroll
      for (int kk = 0; kk < 8; ++kk) Ah[kk] = frag_ld(hhb, (size_t)mrow, 256, kk * 32, lane);

#pragma unroll 1
      for (int s6 = 0; s6 < 6; ++s6) {
        int s = w * 6 + s6;                 // scalar subtile index
        int n0 = s << 4;
        // Issue ALL B-fragment loads for this subtile first (one big clause);
        // in-order load returns let WMMA k-step i wait only for load i.
        v16bf Bi[2], Bh[8];
#pragma unroll
        for (int kk = 0; kk < 2; ++kk)
          Bi[kk] = frag_ld(WihT, (size_t)(n0 + mrow), 64, kk * 32, lane);
#pragma unroll
        for (int kk = 0; kk < 8; ++kk)
          Bh[kk] = frag_ld(WhhT, (size_t)(n0 + mrow), 256, kk * 32, lane);

        v8f ai = {0.f, 0.f, 0.f, 0.f, 0.f, 0.f, 0.f, 0.f};
        v8f ah = {0.f, 0.f, 0.f, 0.f, 0.f, 0.f, 0.f, 0.f};
#pragma unroll
        for (int kk = 0; kk < 2; ++kk)
          ai = __builtin_amdgcn_wmma_f32_16x16x32_bf16(false, Ai[kk], false, Bi[kk],
                                                       (short)0, ai, false, false);
#pragma unroll
        for (int kk = 0; kk < 8; ++kk)
          ah = __builtin_amdgcn_wmma_f32_16x16x32_bf16(false, Ah[kk], false, Bh[kk],
                                                       (short)0, ah, false, false);

        // C/D layout: lane l, acc[r] -> row = r + 8*(l>=16), col = n0 + (l&15)
        int lm = (lane >> 4) << 3;
        int nc = n0 + mrow;
        if (s < 32) {                       // scalar branch (w, s6 uniform)
#pragma unroll
          for (int r = 0; r < 8; ++r) Grz[(lm + r) * 512 + nc] = ai[r] + ah[r];
        } else {
          int c = nc - 512;
#pragma unroll
          for (int r = 0; r < 8; ++r) {
            Gni[(lm + r) * 256 + c] = ai[r];
            Gnh[(lm + r) * 256 + c] = ah[r];
          }
        }
      }
    }
    __syncthreads();

    // ---- phase 3: gates + state update + sel capture ----
    {
      int lt = llen[mthr];
#pragma unroll 4
      for (int kk = 0; kk < 16; ++kk) {
        int k = cthr + kk;
        float gr = Grz[mthr * 512 + k]       + bias[256 + k]  + bias[1024 + k];
        float gz = Grz[mthr * 512 + 256 + k] + bias[512 + k]  + bias[1280 + k];
        float rr = 1.f / (1.f + __expf(-gr));
        float zz = 1.f / (1.f + __expf(-gz));
        float nn = tanhf(Gni[mthr * 256 + k] + bias[768 + k] +
                         rr * (Gnh[mthr * 256 + k] + bias[1536 + k]));
        float hv = (1.f - zz) * nn + zz * hhf[mthr * 256 + k];
        h[mthr * 256 + k] = hv;
        if (t == lt) sel[(size_t)(b0 + mthr) * Hd + k] = hv;
      }
    }
    __syncthreads();
  }
}

// ---------------------------------------------------------------------------
// Kernel 5: out = sigmoid([sel, age, gender] @ fc_W^T + fc_b)
__global__ void k_fc(const float* __restrict__ sel, const float* __restrict__ age,
                     const float* __restrict__ gender, const float* __restrict__ fcW,
                     const float* __restrict__ fcb, float* __restrict__ out) {
  int b = blockIdx.x * 256 + threadIdx.x;
  if (b >= Bsz) return;
  float s = fcb[0];
  const float* sr = sel + (size_t)b * Hd;
  for (int k = 0; k < Hd; ++k) s += sr[k] * fcW[k];
  s += age[b] * fcW[Hd] + gender[b] * fcW[Hd + 1];
  out[b] = 1.f / (1.f + __expf(-s));
}

// ---------------------------------------------------------------------------
extern "C" void kernel_launch(void* const* d_in, const int* in_sizes, int n_in,
                              void* d_out, int out_size, void* d_ws, size_t ws_size,
                              hipStream_t stream) {
  const float* x      = (const float*)d_in[0];
  const float* h0     = (const float*)d_in[1];
  const float* mm     = (const float*)d_in[2];
  const float* d      = (const float*)d_in[3];
  const float* x_m    = (const float*)d_in[4];
  const float* age    = (const float*)d_in[5];
  const float* gender = (const float*)d_in[6];
  const int*   length = (const int*)d_in[7];
  const float* dec_w  = (const float*)d_in[8];
  const float* dec_b  = (const float*)d_in[9];
  const float* initW  = (const float*)d_in[10];
  const float* initb  = (const float*)d_in[11];
  const float* hdW    = (const float*)d_in[12];
  const float* hdb    = (const float*)d_in[13];
  const float* Wih    = (const float*)d_in[14];
  const float* Whh    = (const float*)d_in[15];
  const float* bih    = (const float*)d_in[16];
  const float* bhh    = (const float*)d_in[17];
  const float* fcW    = (const float*)d_in[18];
  const float* fcb    = (const float*)d_in[19];
  float* out = (float*)d_out;

  // Workspace layout (~69 MB)
  char* ws = (char*)d_ws;
  __hip_bfloat16* XT   = (__hip_bfloat16*)ws; ws += (size_t)Bsz * Tw * 64 * 2;
  __hip_bfloat16* Wihp = (__hip_bfloat16*)ws; ws += (size_t)H3 * 64 * 2;
  __hip_bfloat16* Whhb = (__hip_bfloat16*)ws; ws += (size_t)H3 * Hd * 2;
  float* Wbar = (float*)ws;                   ws += (size_t)Fin * Tw * 4;
  float* bbar = (float*)ws;                   ws += 256;
  float* sel  = (float*)ws;                   ws += (size_t)Bsz * Hd * 4;

  k_wbar <<<(Fin * Tw + 255) / 256, 256, 0, stream>>>(initW, initb, Wbar, bbar);
  k_attn <<<Bsz, 256, 0, stream>>>(x, mm, d, x_m, dec_w, dec_b, Wbar, bbar, XT);
  k_convw<<<(H3 * 64 + H3 * Hd + 255) / 256, 256, 0, stream>>>(Wih, Whh, Wihp, Whhb);
  k_scan <<<Bsz / 16, 256, SCAN_SMEM_BYTES, stream>>>(XT, Wihp, Whhb, d, hdW, hdb,
                                                      bih, bhh, h0, length, sel);
  k_fc   <<<(Bsz + 255) / 256, 256, 0, stream>>>(sel, age, gender, fcW, fcb, out);
}
